// GraphAttentionLayer_45011257262225
// MI455X (gfx1250) — matrix operand order repaired and checked by reading, not compiled, not run on hardware
//
#include <hip/hip_runtime.h>

typedef __attribute__((ext_vector_type(16))) _Float16 v16h;
typedef __attribute__((ext_vector_type(8)))  float    v8f;

#define N_NODES  8192
#define IN_FEAT  512
#define OUT_FEAT 128
#define ALPHA    0.2f
#define RND_SCALE 1e5f

union HFrag {
  v16h v;
  _Float16 e[16];
  unsigned int u[8];
};

// lane gather via DS (safe on gfx1250 wave32; avoids wavefront-size macro issues)
__device__ __forceinline__ float lane_get(float x, int srcLane) {
  return __builtin_bit_cast(float,
      __builtin_amdgcn_ds_bpermute(srcLane << 2, __builtin_bit_cast(int, x)));
}

// ---------------------------------------------------------------------------
// k0: W [512x128] f32  ->  Wp f16 packed as (k-pair, n): u32 = {f16(W[2p,n]), f16(W[2p+1,n])}
// ---------------------------------------------------------------------------
__global__ void k0_convW(const float* __restrict__ W, unsigned int* __restrict__ Wp) {
  int tid = blockIdx.x * blockDim.x + threadIdx.x;          // 0 .. 512*128/2-1
  int pair = tid / OUT_FEAT;
  int n    = tid % OUT_FEAT;
  int k0   = pair * 2;
  _Float16 lo = (_Float16)W[k0 * OUT_FEAT + n];
  _Float16 hi = (_Float16)W[(k0 + 1) * OUT_FEAT + n];
  unsigned int ulo = __builtin_bit_cast(unsigned short, lo);
  unsigned int uhi = __builtin_bit_cast(unsigned short, hi);
  Wp[tid] = ulo | (uhi << 16);
}

// ---------------------------------------------------------------------------
// k1: Wh = h @ W   (f16 WMMA, f32 accum). One wave per 16 rows.
// Writes Wh fp32 [N,128] and Whp: f16 row-paired layout
//   Whp_half[(row>>1)*256 + n*2 + (row&1)]  (so (row,row+1) at col n are one u32)
// ---------------------------------------------------------------------------
__global__ void k1_gemm_hW(const float* __restrict__ h,
                           const unsigned int* __restrict__ Wp,
                           float* __restrict__ Wh,
                           unsigned short* __restrict__ Whp) {
  const int lane = threadIdx.x & 31;
  const int m    = lane & 15;
  const int hs   = lane >> 4;            // lane half: selects K sub-range
  const int i0   = blockIdx.x * 16;
  const int row  = i0 + m;

  const v8f vzero = {0.f,0.f,0.f,0.f,0.f,0.f,0.f,0.f};
  v8f acc[8];
#pragma unroll
  for (int f = 0; f < 8; ++f) acc[f] = vzero;

  for (int kc = 0; kc < IN_FEAT; kc += 32) {
    // A fragment: 16x32 f16, contiguous layout (lanes 0-15: K=0..15 ; lanes 16-31: K=16..31)
    HFrag af;
#pragma unroll
    for (int v = 0; v < 8; ++v) {
      int kk = 2 * v + 16 * hs;
      const float2 p = *(const float2*)(h + row * IN_FEAT + kc + kk);
      af.e[2 * v]     = (_Float16)p.x;
      af.e[2 * v + 1] = (_Float16)p.y;
    }
    // B fragments (32x16 each) from packed Wp, one b32 per VGPR
#pragma unroll
    for (int f = 0; f < 8; ++f) {
      HFrag bf;
      const int n = f * 16 + m;
#pragma unroll
      for (int v = 0; v < 8; ++v) {
        int pairidx = (kc >> 1) + v + 8 * hs;     // K = kc + 2v + 16*hs
        bf.u[v] = Wp[pairidx * OUT_FEAT + n];
      }
      acc[f] = __builtin_amdgcn_wmma_f32_16x16x32_f16(
          false, af.v, false, bf.v, (short)0, acc[f], false, false);
    }
  }

  // C/D layout: VGPR r: lanes 0-15 -> M=r, lanes 16-31 -> M=r+8 ; N = f*16 + m
#pragma unroll
  for (int f = 0; f < 8; ++f) {
    const int n = f * 16 + m;
#pragma unroll
    for (int r = 0; r < 8; ++r) {
      const int grow = i0 + r + 8 * hs;
      const float val = acc[f][r];
      Wh[grow * OUT_FEAT + n] = val;
      _Float16 hv = (_Float16)val;
      Whp[(grow >> 1) * (2 * OUT_FEAT) + n * 2 + (grow & 1)] =
          __builtin_bit_cast(unsigned short, hv);
    }
  }
}

// ---------------------------------------------------------------------------
// k2: Wh1[i] = Wh[i,:]·a[0:128], Wh2[i] = Wh[i,:]·a[128:256]. Wave per row.
// ---------------------------------------------------------------------------
__global__ void k2_proj(const float* __restrict__ Wh, const float* __restrict__ a,
                        float* __restrict__ Wh1, float* __restrict__ Wh2) {
  const int lane = threadIdx.x & 31;
  const int rowi = blockIdx.x * (blockDim.x >> 5) + (threadIdx.x >> 5);
  const float4 w  = *(const float4*)(Wh + rowi * OUT_FEAT + lane * 4);
  const float4 a1 = *(const float4*)(a + lane * 4);
  const float4 a2 = *(const float4*)(a + OUT_FEAT + lane * 4);
  float s1 = w.x * a1.x + w.y * a1.y + w.z * a1.z + w.w * a1.w;
  float s2 = w.x * a2.x + w.y * a2.y + w.z * a2.z + w.w * a2.w;
#pragma unroll
  for (int off = 16; off > 0; off >>= 1) {
    s1 += lane_get(s1, lane ^ off);
    s2 += lane_get(s2, lane ^ off);
  }
  if (lane == 0) { Wh1[rowi] = s1; Wh2[rowi] = s2; }
}

// ---------------------------------------------------------------------------
// k3: fused  e = lrelu(Wh1_i + Wh2_j)  ->  mask(adj)  ->  online softmax
//            ->  acc += P @ Wh  (f16 WMMA)  ->  (+rnd)/1e5 -> ELU
// One wave per 16 output rows; sweeps j in chunks of 32.
// ---------------------------------------------------------------------------
__global__ void k3_attn(const int* __restrict__ adj,
                        const unsigned int* __restrict__ Whp,   // u32 = 2 packed f16
                        const float* __restrict__ Wh1,
                        const float* __restrict__ Wh2,
                        const float* __restrict__ rnd,
                        float* __restrict__ out) {
  const int lane = threadIdx.x & 31;
  const int m    = lane & 15;
  const int hs   = lane >> 4;
  const int i0   = blockIdx.x * 16;
  const int row  = i0 + m;               // score row owned by this lane
  const float wh1 = Wh1[row];
  const size_t rowbase = (size_t)row * N_NODES;

  const v8f vzero = {0.f,0.f,0.f,0.f,0.f,0.f,0.f,0.f};
  v8f acc[8];
#pragma unroll
  for (int f = 0; f < 8; ++f) acc[f] = vzero;
  float mrow = -3.0e38f;   // running max for row (l%16); identical in lanes l, l+16
  float lsum = 0.0f;       // running denom

  for (int j0 = 0; j0 < N_NODES; j0 += 32) {
    // --- scores for this lane's 16 (row, j) elements (A-fragment ordering) ---
    float s[16]; int mk[16];
    float tmax = -3.0e38f;
#pragma unroll
    for (int v = 0; v < 8; ++v) {
      const int jj = j0 + 2 * v + 16 * hs;
      const float2 w2 = *(const float2*)(Wh2 + jj);
      const int2   ad = *(const int2*)(adj + rowbase + jj);
      float s0 = wh1 + w2.x; s0 = (s0 > 0.f) ? s0 : ALPHA * s0;
      float s1 = wh1 + w2.y; s1 = (s1 > 0.f) ? s1 : ALPHA * s1;
      s[2 * v] = s0;  s[2 * v + 1] = s1;
      mk[2 * v] = ad.x; mk[2 * v + 1] = ad.y;
      tmax = fmaxf(tmax, (ad.x > 0) ? s0 : -3.0e38f);
      tmax = fmaxf(tmax, (ad.y > 0) ? s1 : -3.0e38f);
    }
    tmax = fmaxf(tmax, lane_get(tmax, lane ^ 16));   // combine j-halves of row
    const float mnew = fmaxf(mrow, tmax);
    const float c = __expf(mrow - mnew);             // rescale of old state

    // --- probabilities -> f16 A fragment + tile row-sum ---
    HFrag pf;
    float tsum = 0.f;
#pragma unroll
    for (int e = 0; e < 16; ++e) {
      float p = (mk[e] > 0) ? __expf(s[e] - mnew) : 0.0f;
      tsum += p;
      pf.e[e] = (_Float16)p;
    }
    tsum += lane_get(tsum, lane ^ 16);
    lsum = lsum * c + tsum;
    mrow = mnew;

    // --- rescale accumulators (C rows are r + 8*hs, stats live at lane r+8*hs) ---
    float cr[8];
#pragma unroll
    for (int r = 0; r < 8; ++r) cr[r] = lane_get(c, r + 8 * hs);
#pragma unroll
    for (int f = 0; f < 8; ++f)
#pragma unroll
      for (int r = 0; r < 8; ++r) acc[f][r] *= cr[r];

    // --- B fragments from L2-resident packed Wh and WMMA accumulate ---
    const int pbase = (j0 >> 1) + 8 * hs;            // K = j0 + 2v + 16*hs
#pragma unroll
    for (int f = 0; f < 8; ++f) {
      HFrag bf;
      const int n = f * 16 + m;
#pragma unroll
      for (int v = 0; v < 8; ++v) bf.u[v] = Whp[(pbase + v) * OUT_FEAT + n];
      acc[f] = __builtin_amdgcn_wmma_f32_16x16x32_f16(
          false, pf.v, false, bf.v, (short)0, acc[f], false, false);
    }
  }

  // --- epilogue: normalize, add rnd, /1e5, ELU ---
  float lr[8];
#pragma unroll
  for (int r = 0; r < 8; ++r) lr[r] = lane_get(lsum, r + 8 * hs);
#pragma unroll
  for (int f = 0; f < 8; ++f) {
    const int n = f * 16 + m;
#pragma unroll
    for (int r = 0; r < 8; ++r) {
      const int grow = i0 + r + 8 * hs;
      float v = acc[f][r] / fmaxf(lr[r], 1e-30f);
      v = (v + rnd[grow * OUT_FEAT + n]) * (1.0f / RND_SCALE);
      out[grow * OUT_FEAT + n] = (v > 0.f) ? v : (__expf(v) - 1.0f);
    }
  }
}

// ---------------------------------------------------------------------------
extern "C" void kernel_launch(void* const* d_in, const int* in_sizes, int n_in,
                              void* d_out, int out_size, void* d_ws, size_t ws_size,
                              hipStream_t stream) {
  const float* h   = (const float*)d_in[0];
  const int*   adj = (const int*)d_in[1];
  const float* W   = (const float*)d_in[2];
  const float* a   = (const float*)d_in[3];
  const float* rnd = (const float*)d_in[4];
  float* out = (float*)d_out;

  char* ws = (char*)d_ws;
  float*          Wh  = (float*)ws;                                   // 4 MB
  unsigned short* Whp = (unsigned short*)(ws + 4u * 1024 * 1024);     // 2 MB
  unsigned int*   Wp  = (unsigned int*)(ws + 6u * 1024 * 1024);       // 128 KB
  float*          Wh1 = (float*)(ws + 6u * 1024 * 1024 + 128u * 1024);
  float*          Wh2 = Wh1 + N_NODES;

  k0_convW <<<(IN_FEAT * OUT_FEAT / 2) / 256, 256, 0, stream>>>(W, Wp);
  k1_gemm_hW<<<N_NODES / 16, 32, 0, stream>>>(h, Wp, Wh, Whp);
  k2_proj  <<<N_NODES / 8, 256, 0, stream>>>(Wh, a, Wh1, Wh2);
  k3_attn  <<<N_NODES / 16, 32, 0, stream>>>(adj, (const unsigned int*)Whp,
                                             Wh1, Wh2, rnd, out);
}
